// BertDeAttention_85676007621155
// MI455X (gfx1250) — matrix operand order repaired and checked by reading, not compiled, not run on hardware
//
#include <hip/hip_runtime.h>
#include <hip/hip_bf16.h>
#include <cstdint>

// Problem constants (match reference)
#define B_    8
#define L_    1024
#define HID_  1024
#define NH_   16
#define DH_   64

typedef __attribute__((ext_vector_type(16))) _Float16 v16h;
typedef __attribute__((ext_vector_type(8)))  float    v8f;

union FragU { v16h v; uint4 q[2]; };

// ---------------------------------------------------------------------------
// WMMA fragment loaders (CDNA5 ISA §7.12.2 layouts), wave32.
// Source memory is K-contiguous with row stride `ld` (in halves).
// ---------------------------------------------------------------------------

// A-matrix 16x32 f16: lanes 0-15 row M=lane, K = {0..7,16..23};
// lanes 16-31 row M=lane-16, K = {8..15,24..31}.
__device__ __forceinline__ v16h load_frag_a(const _Float16* base, int ld) {
  const int lane = threadIdx.x & 31;
  const int row  = lane & 15;
  const int kh   = (lane >> 4) << 3;            // 0 or 8
  const _Float16* p = base + (size_t)row * ld + kh;
  FragU f;
  f.q[0] = *reinterpret_cast<const uint4*>(p);        // K = kh .. kh+7
  f.q[1] = *reinterpret_cast<const uint4*>(p + 16);   // K = kh+16 .. kh+23
  return f.v;
}

// B-matrix 32x16 f16: lanes 0-15 col N=lane, K = {0..7,16..23};
// lanes 16-31 col N=lane-16, K = {8..15,24..31}.
__device__ __forceinline__ v16h load_frag_b(const _Float16* base, int ld) {
  const int lane = threadIdx.x & 31;
  const int col  = lane & 15;
  const int kh   = (lane >> 4) << 3;            // 0 or 8
  const _Float16* p = base + (size_t)col * ld + kh;
  FragU f;
  f.q[0] = *reinterpret_cast<const uint4*>(p);        // K = kh .. kh+7
  f.q[1] = *reinterpret_cast<const uint4*>(p + 16);   // K = kh+16 .. kh+23
  return f.v;
}

__device__ __forceinline__ v8f wmma16(v16h a, v16h b, v8f c) {
  return __builtin_amdgcn_wmma_f32_16x16x32_f16(false, a, false, b,
                                                (short)0, c, false, false);
}

// ds_swizzle_b32 XOR butterfly (ISA §12.12 op 53, group-of-32 mode):
// offset[15]=0, and_mask=0x1f, or_mask=0, xor_mask=XOR. Masks 1/2/4/8 stay
// within each 16-lane half (rows of a WMMA C tile), matching shfl_xor(.,16).
template <int XOR>
__device__ __forceinline__ float swz_xor(float v) {
  int r = __builtin_amdgcn_ds_swizzle(__float_as_int(v), (XOR << 10) | 0x1f);
  return __int_as_float(r);
}

__device__ __forceinline__ float red_max16(float v) {
  v = fmaxf(v, swz_xor<1>(v));
  v = fmaxf(v, swz_xor<2>(v));
  v = fmaxf(v, swz_xor<4>(v));
  v = fmaxf(v, swz_xor<8>(v));
  return v;
}
__device__ __forceinline__ float red_sum16(float v) {
  v += swz_xor<1>(v);
  v += swz_xor<2>(v);
  v += swz_xor<4>(v);
  v += swz_xor<8>(v);
  return v;
}

// ---------------------------------------------------------------------------
// fp32 -> fp16 conversion (vectorized: 4 elems / thread)
// ---------------------------------------------------------------------------
__global__ __launch_bounds__(256) void cvt_f16_kernel(
    const float* __restrict__ in, _Float16* __restrict__ out, int n4) {
  int i = blockIdx.x * 256 + threadIdx.x;
  if (i < n4) {
    float4 f = reinterpret_cast<const float4*>(in)[i];
    alignas(8) _Float16 h[4] = {(_Float16)f.x, (_Float16)f.y,
                                (_Float16)f.z, (_Float16)f.w};
    reinterpret_cast<unsigned long long*>(out)[i] =
        *reinterpret_cast<unsigned long long*>(h);
  }
}

// ---------------------------------------------------------------------------
// Register-blocked GEMM: out = A[8192][1024] * W[1024][1024]^T + bias
// Each wave computes a 64x32 output block (4 M-tiles x 2 N-tiles):
//   per K-step: 4 A-frags + 2 B-frags -> 8 WMMAs (24 B/lane per WMMA).
// TRANSV: store per-head transposed Vt[(b*HID + col)*L + l] (col = h*DH+d).
// OutT: _Float16 (projections) or float (output dense, pre-LN).
// ---------------------------------------------------------------------------
template <bool TRANSV, typename OutT>
__global__ __launch_bounds__(128) void gemm16_kernel(
    const _Float16* __restrict__ A, const _Float16* __restrict__ W,
    const float* __restrict__ bias, OutT* __restrict__ out) {
  const int warp = threadIdx.x >> 5;
  const int lane = threadIdx.x & 31;
  const int tm   = (blockIdx.y * 4 + warp) * 4;   // first of 4 M tiles (0..508)
  const int tn   = blockIdx.x * 2;                // first of 2 N tiles (0..62)
  const _Float16* a0 = A + (size_t)tm * 16 * HID_;
  const _Float16* w0 = W + (size_t)tn * 16 * HID_;

  v8f acc[4][2] = {};
  for (int k0 = 0; k0 < HID_; k0 += 32) {
    const v16h bf0 = load_frag_b(w0 + k0, HID_);
    const v16h bf1 = load_frag_b(w0 + (size_t)16 * HID_ + k0, HID_);
#pragma unroll
    for (int i = 0; i < 4; ++i) {
      const v16h af = load_frag_a(a0 + (size_t)i * 16 * HID_ + k0, HID_);
      acc[i][0] = wmma16(af, bf0, acc[i][0]);
      acc[i][1] = wmma16(af, bf1, acc[i][1]);
    }
  }

  const int n  = lane & 15;
  const int mb = (lane >> 4) << 3;
#pragma unroll
  for (int c = 0; c < 2; ++c) {
    const int col = (tn + c) * 16 + n;
    const float bz = bias[col];
#pragma unroll
    for (int i = 0; i < 4; ++i) {
      const int rowBase = (tm + i) * 16 + mb;
#pragma unroll
      for (int j = 0; j < 8; ++j) {
        const float v = acc[i][c][j] + bz;
        const int row = rowBase + j;
        if (TRANSV) {
          const int b = row >> 10, l = row & (L_ - 1);
          out[((size_t)(b * HID_ + col)) * L_ + l] = (OutT)v;
        } else {
          out[(size_t)row * HID_ + col] = (OutT)v;
        }
      }
    }
  }
}

// ---------------------------------------------------------------------------
// Flash attention: one wave per (b, h, 16-query tile). 64-key chunks:
//   S = Q(16x64) . K_chunk^T -> four 16x16 WMMA C tiles (2 k-steps each)
//   online softmax: per-lane local reduce over 4 tiles, then one 4-step
//   ds_swizzle XOR butterfly per row (amortized: 64 ds ops per 16 WMMAs)
//   P (16x64) staged through per-wave LDS tile -> two A-fragments
//   ctx += P . V_chunk(64x64) -> 8 WMMAs (Vt is keys-contiguous)
// ---------------------------------------------------------------------------
__global__ __launch_bounds__(128) void flash_attn_kernel(
    const _Float16* __restrict__ Q,   // [B*L][HID] (b,l,h,d)
    const _Float16* __restrict__ Km,  // [B*L][HID]
    const _Float16* __restrict__ Vt,  // [B*HID][L] per-head transposed
    const float* __restrict__ mask,   // [B][L] additive
    _Float16* __restrict__ ctx) {     // [B*L][HID]
  __shared__ alignas(16) _Float16 ldsP[4][16 * 80];  // 80-half row stride
  const int warp = threadIdx.x >> 5;
  const int lane = threadIdx.x & 31;
  const int n    = lane & 15;
  const int mb   = (lane >> 4) << 3;
  const int qb   = (blockIdx.x * 4 + warp) * 16;
  const int h    = blockIdx.y;
  const int b    = blockIdx.z;

  const _Float16* qptr = Q + ((size_t)(b * L_) + qb) * HID_ + h * DH_;
  const v16h aq0 = load_frag_a(qptr, HID_);        // d 0..31
  const v16h aq1 = load_frag_a(qptr + 32, HID_);   // d 32..63

  float mstate[8], lstate[8];
  v8f acc0 = {}, acc1 = {}, acc2 = {}, acc3 = {};
#pragma unroll
  for (int j = 0; j < 8; ++j) { mstate[j] = -3.0e30f; lstate[j] = 0.0f; }

  const float* mrow = mask + (size_t)b * L_;
  _Float16* Pw = &ldsP[warp][0];
  const _Float16* Pr = &ldsP[warp][(lane & 15) * 80 + ((lane >> 4) << 3)];

  for (int kc = 0; kc < L_; kc += 64) {
    const _Float16* kptr = Km + ((size_t)(b * L_) + kc) * HID_ + h * DH_;
    v8f s[4];
#pragma unroll
    for (int t = 0; t < 4; ++t) {
      v8f z = {};
      z = wmma16(aq0, load_frag_b(kptr + (size_t)(t * 16) * HID_, HID_), z);
      z = wmma16(aq1, load_frag_b(kptr + (size_t)(t * 16) * HID_ + 32, HID_), z);
      s[t] = z;
    }
    float mk[4];
#pragma unroll
    for (int t = 0; t < 4; ++t) mk[t] = mrow[kc + t * 16 + n];

#pragma unroll
    for (int j = 0; j < 8; ++j) {
      float a[4];
#pragma unroll
      for (int t = 0; t < 4; ++t) a[t] = s[t][j] * 0.125f + mk[t];
      const float vmax =
          red_max16(fmaxf(fmaxf(a[0], a[1]), fmaxf(a[2], a[3])));
      const float mnew  = fmaxf(mstate[j], vmax);
      const float alpha = __expf(mstate[j] - mnew);
      float p[4], rs = 0.0f;
#pragma unroll
      for (int t = 0; t < 4; ++t) { p[t] = __expf(a[t] - mnew); rs += p[t]; }
      rs = red_sum16(rs);
      lstate[j] = lstate[j] * alpha + rs;
      mstate[j] = mnew;
      acc0[j] *= alpha; acc1[j] *= alpha; acc2[j] *= alpha; acc3[j] *= alpha;
#pragma unroll
      for (int t = 0; t < 4; ++t) s[t][j] = p[t];
    }

    // C-layout -> A-layout: bounce P (16q x 64keys) through per-wave LDS
#pragma unroll
    for (int j = 0; j < 8; ++j) {
#pragma unroll
      for (int t = 0; t < 4; ++t)
        Pw[(mb + j) * 80 + t * 16 + n] = (_Float16)s[t][j];
    }
    // same-wave LDS RAW: compiler inserts s_wait_dscnt
    FragU pf0, pf1;
    pf0.q[0] = *reinterpret_cast<const uint4*>(Pr);
    pf0.q[1] = *reinterpret_cast<const uint4*>(Pr + 16);
    pf1.q[0] = *reinterpret_cast<const uint4*>(Pr + 32);
    pf1.q[1] = *reinterpret_cast<const uint4*>(Pr + 48);

    const _Float16* vptr = Vt + ((size_t)(b * HID_) + h * DH_) * L_ + kc;
    acc0 = wmma16(pf0.v, load_frag_b(vptr,                L_), acc0);
    acc0 = wmma16(pf1.v, load_frag_b(vptr + 32,           L_), acc0);
    acc1 = wmma16(pf0.v, load_frag_b(vptr + 16 * L_,      L_), acc1);
    acc1 = wmma16(pf1.v, load_frag_b(vptr + 16 * L_ + 32, L_), acc1);
    acc2 = wmma16(pf0.v, load_frag_b(vptr + 32 * L_,      L_), acc2);
    acc2 = wmma16(pf1.v, load_frag_b(vptr + 32 * L_ + 32, L_), acc2);
    acc3 = wmma16(pf0.v, load_frag_b(vptr + 48 * L_,      L_), acc3);
    acc3 = wmma16(pf1.v, load_frag_b(vptr + 48 * L_ + 32, L_), acc3);
  }

  _Float16* optr = ctx + ((size_t)(b * L_) + qb + mb) * HID_ + h * DH_ + n;
#pragma unroll
  for (int j = 0; j < 8; ++j) {
    const float inv = 1.0f / lstate[j];
    optr[(size_t)j * HID_ + 0]  = (_Float16)(acc0[j] * inv);
    optr[(size_t)j * HID_ + 16] = (_Float16)(acc1[j] * inv);
    optr[(size_t)j * HID_ + 32] = (_Float16)(acc2[j] * inv);
    optr[(size_t)j * HID_ + 48] = (_Float16)(acc3[j] * inv);
  }
}

// ---------------------------------------------------------------------------
// Residual add + LayerNorm: out = gamma * (x - mu) * rsqrt(var+eps) + beta
// x = dense + resid; one block (256 thr) per 1024-elem row.
// ---------------------------------------------------------------------------
__global__ __launch_bounds__(256) void resid_ln_kernel(
    const float* __restrict__ dense, const float* __restrict__ resid,
    const float* __restrict__ gamma, const float* __restrict__ beta,
    float* __restrict__ out) {
  const int row = blockIdx.x;
  const int t   = threadIdx.x;
  const float* dr = dense + (size_t)row * HID_;
  const float* rr = resid + (size_t)row * HID_;
  float x[4];
  float s1 = 0.0f, s2 = 0.0f;
#pragma unroll
  for (int i = 0; i < 4; ++i) {
    float v = dr[t + i * 256] + rr[t + i * 256];
    x[i] = v; s1 += v; s2 += v * v;
  }
  __shared__ float red1[256];
  __shared__ float red2[256];
  red1[t] = s1; red2[t] = s2;
  __syncthreads();
  for (int off = 128; off > 0; off >>= 1) {
    if (t < off) { red1[t] += red1[t + off]; red2[t] += red2[t + off]; }
    __syncthreads();
  }
  const float mu  = red1[0] * (1.0f / HID_);
  const float var = red2[0] * (1.0f / HID_) - mu * mu;
  const float rs  = rsqrtf(var + 1e-12f);
#pragma unroll
  for (int i = 0; i < 4; ++i) {
    int c = t + i * 256;
    out[(size_t)row * HID_ + c] = gamma[c] * (x[i] - mu) * rs + beta[c];
  }
}

// ---------------------------------------------------------------------------
// Host launcher
// ---------------------------------------------------------------------------
extern "C" void kernel_launch(void* const* d_in, const int* in_sizes, int n_in,
                              void* d_out, int out_size, void* d_ws,
                              size_t ws_size, hipStream_t stream) {
  const float* cin   = (const float*)d_in[0];
  const float* qin   = (const float*)d_in[1];
  const float* maskc = (const float*)d_in[2];
  const float* maskq = (const float*)d_in[3];
  const float* Wsrc[7] = {(const float*)d_in[4],  (const float*)d_in[6],
                          (const float*)d_in[8],  (const float*)d_in[10],
                          (const float*)d_in[12], (const float*)d_in[14],
                          (const float*)d_in[16]};
  const float* bq  = (const float*)d_in[5];
  const float* bk  = (const float*)d_in[7];
  const float* bv  = (const float*)d_in[9];
  const float* bqq = (const float*)d_in[11];
  const float* bqk = (const float*)d_in[13];
  const float* bqv = (const float*)d_in[15];
  const float* bo  = (const float*)d_in[17];
  const float* gamma = (const float*)d_in[18];
  const float* beta  = (const float*)d_in[19];
  float* out = (float*)d_out;

  char* ws = (char*)d_ws;
  size_t off = 0;
  auto carve = [&](size_t bytes) -> char* {
    char* p = ws + off;
    off += (bytes + 255) & ~(size_t)255;
    return p;
  };
  const size_t NTOK = (size_t)B_ * L_;   // 8192 rows
  const size_t EH   = NTOK * HID_;       // 8M elements

  _Float16* enc16 = (_Float16*)carve(EH * 2);
  _Float16* dec16 = (_Float16*)carve(EH * 2);
  _Float16* w16[7];
  for (int i = 0; i < 7; ++i) w16[i] = (_Float16*)carve((size_t)HID_ * HID_ * 2);
  _Float16* Qc   = (_Float16*)carve(EH * 2);
  _Float16* Kc   = (_Float16*)carve(EH * 2);
  _Float16* Vct  = (_Float16*)carve(EH * 2);
  _Float16* Qq   = (_Float16*)carve(EH * 2);
  _Float16* Kq   = (_Float16*)carve(EH * 2);
  _Float16* Vqt  = (_Float16*)carve(EH * 2);
  _Float16* ctxc = (_Float16*)carve(EH * 2);
  _Float16* ctxq = (_Float16*)carve(EH * 2);
  float* densec  = (float*)carve(EH * 4);
  float* denseq  = (float*)carve(EH * 4);

  // ---- fp32 -> fp16 conversions
  const int n4 = (int)(EH / 4);                    // 2M
  cvt_f16_kernel<<<n4 / 256, 256, 0, stream>>>(cin, enc16, n4);
  cvt_f16_kernel<<<n4 / 256, 256, 0, stream>>>(qin, dec16, n4);
  const int w4 = HID_ * HID_ / 4;                  // 256K
  for (int i = 0; i < 7; ++i)
    cvt_f16_kernel<<<w4 / 256, 256, 0, stream>>>(Wsrc[i], w16[i], w4);

  // ---- QKV projections (c-branch: Q<-dec, K/V<-enc; q-branch: mirrored)
  // wave = 64x32 output block; block = 256x32; grid = (N/32, M/256)
  dim3 gg(HID_ / 32, (unsigned)(NTOK / 256));      // (32, 32), 128 thr
  gemm16_kernel<false, _Float16><<<gg, 128, 0, stream>>>(dec16, w16[0], bq,  Qc);
  gemm16_kernel<false, _Float16><<<gg, 128, 0, stream>>>(enc16, w16[1], bk,  Kc);
  gemm16_kernel<true,  _Float16><<<gg, 128, 0, stream>>>(enc16, w16[2], bv,  Vct);
  gemm16_kernel<false, _Float16><<<gg, 128, 0, stream>>>(enc16, w16[3], bqq, Qq);
  gemm16_kernel<false, _Float16><<<gg, 128, 0, stream>>>(dec16, w16[4], bqk, Kq);
  gemm16_kernel<true,  _Float16><<<gg, 128, 0, stream>>>(dec16, w16[5], bqv, Vqt);

  // ---- flash attention, both branches
  dim3 gfa(L_ / 64, NH_, B_);                      // (16, 16, 8), 128 thr
  flash_attn_kernel<<<gfa, 128, 0, stream>>>(Qc, Kc, Vct, maskc, ctxc);
  flash_attn_kernel<<<gfa, 128, 0, stream>>>(Qq, Kq, Vqt, maskq, ctxq);

  // ---- output dense (shared Wo/bo), f32 out
  gemm16_kernel<false, float><<<gg, 128, 0, stream>>>(ctxc, w16[6], bo, densec);
  gemm16_kernel<false, float><<<gg, 128, 0, stream>>>(ctxq, w16[6], bo, denseq);

  // ---- residual (+cinput for BOTH, per reference) + LayerNorm
  resid_ln_kernel<<<(int)NTOK, 256, 0, stream>>>(densec, cin, gamma, beta, out);
  resid_ln_kernel<<<(int)NTOK, 256, 0, stream>>>(denseq, cin, gamma, beta,
                                                 out + EH);
}
